// GSRepair_39324720562556
// MI455X (gfx1250) — matrix operand order repaired and checked by reading, not compiled, not run on hardware
//
#include <hip/hip_runtime.h>
#include <math.h>

typedef __attribute__((ext_vector_type(16))) _Float16 v16h;
typedef __attribute__((ext_vector_type(8)))  float    v8f;

// Problem shape (fixed by reference setup_inputs):
#define B_BATCH 2
#define HI 48
#define WI 48
#define NPB (HI*WI)            // 2304 gaussians per batch image
#define NROWS (B_BATCH*NPB)    // 4608 feature rows
#define KPAD 96                // 65 inputs padded to 3 x K32 WMMA steps
#define KSTEPS 3
#define HID 256
#define NCOLS (4*HID)          // 4 heads fused -> 1024 hidden cols
#define MT (NROWS/16)          // 288 M tiles
#define NT (NCOLS/16)          // 64 N tiles
#define HT 96
#define WT 96
#define PIX (HT*WT)
#define CH 256                 // gaussians staged to LDS per chunk

// WMMA f16 fragment swizzles (CDNA5 ISA 16-bit A 16x32 / B 32x16 layouts).
// A element (m,k): lane = m + 16*((k>>3)&1), elem i = (k&7) + 8*((k&31)>>4)
// B element (k,n): lane = n + 16*((k>>3)&1), elem i = (k&7) + 8*((k&31)>>4)
__device__ __forceinline__ int a_frag_index(int mtile, int m, int k) {
  int s = k >> 5, kk = k & 31;
  int lane = m + 16 * ((kk >> 3) & 1);
  int i = (kk & 7) + 8 * (kk >> 4);
  return ((mtile * KSTEPS + s) * 32 + lane) * 16 + i;
}
__device__ __forceinline__ int b_frag_index(int ntile, int k, int n) {
  int s = k >> 5, kk = k & 31;
  int lane = n + 16 * ((kk >> 3) & 1);
  int i = (kk & 7) + 8 * (kk >> 4);
  return ((ntile * KSTEPS + s) * 32 + lane) * 16 + i;
}

// ---------------------------------------------------------------------------
// 1) Pack four 65x256 W1 matrices into f16 B-fragments [64 tiles][3][32][16]
// ---------------------------------------------------------------------------
__global__ __launch_bounds__(256)
void pack_w1_kernel(const float* __restrict__ off_w1, const float* __restrict__ sc_w1,
                    const float* __restrict__ rot_w1, const float* __restrict__ col_w1,
                    const float* __restrict__ off_b1, const float* __restrict__ sc_b1,
                    const float* __restrict__ rot_b1, const float* __restrict__ col_b1,
                    _Float16* __restrict__ w1frag, float* __restrict__ b1cat) {
  int tid = blockIdx.x * blockDim.x + threadIdx.x;
  if (tid >= KPAD * NCOLS) return;
  int k = tid / NCOLS, c = tid % NCOLS;
  int h = c >> 8, cc = c & 255;
  const float* w1 = (h == 0) ? off_w1 : (h == 1) ? sc_w1 : (h == 2) ? rot_w1 : col_w1;
  float v = (k < 65) ? w1[k * HID + cc] : 0.f;
  w1frag[b_frag_index(c >> 4, k, c & 15)] = (_Float16)v;
  if (tid < NCOLS) {
    const float* b1 = (h == 0) ? off_b1 : (h == 1) ? sc_b1 : (h == 2) ? rot_b1 : col_b1;
    b1cat[tid] = b1[cc];
  }
}

// ---------------------------------------------------------------------------
// 2) 3x3 SAME conv (3->64) + factor channel, stored as f16 A-fragments
// ---------------------------------------------------------------------------
__global__ __launch_bounds__(256)
void conv_feat_kernel(const float* __restrict__ inp, const float* __restrict__ enc_w,
                      const float* __restrict__ enc_b, const int* tH, const int* tW,
                      _Float16* __restrict__ featfrag) {
  int tid = blockIdx.x * blockDim.x + threadIdx.x;
  if (tid >= NROWS * KPAD) return;
  int row = tid / KPAD, col = tid % KPAD;
  float v = 0.f;
  if (col < 64) {
    int b = row / NPB, n = row % NPB;
    int y = n / WI, x = n % WI;
    float acc = enc_b[col];
    #pragma unroll
    for (int ci = 0; ci < 3; ++ci) {
      #pragma unroll
      for (int ky = 0; ky < 3; ++ky) {
        int iy = y + ky - 1;
        if (iy < 0 || iy >= HI) continue;
        #pragma unroll
        for (int kx = 0; kx < 3; ++kx) {
          int ix = x + kx - 1;
          if (ix < 0 || ix >= WI) continue;
          acc += inp[((b * 3 + ci) * HI + iy) * WI + ix] *
                 enc_w[((col * 3 + ci) * 3 + ky) * 3 + kx];
        }
      }
    }
    v = acc;
  } else if (col == 64) {
    v = fmaxf((float)tH[0] / (float)HI, (float)tW[0] / (float)WI);
  }
  featfrag[a_frag_index(row >> 4, row & 15, col)] = (_Float16)v;
}

// ---------------------------------------------------------------------------
// 3) Hidden-layer GEMM [4608x96] x [96x1024] -> relu(+bias), via WMMA.
//    Fragments are pre-swizzled: per k-step each lane does two 32B v16h
//    loads (4 x global_load_b128, fully coalesced) + one v_wmma.
// ---------------------------------------------------------------------------
__global__ __launch_bounds__(256)
void mlp_gemm_kernel(const _Float16* __restrict__ featfrag,
                     const _Float16* __restrict__ w1frag,
                     const float* __restrict__ b1cat, float* __restrict__ hidden) {
  const int lane = threadIdx.x & 31;
  const int wave = threadIdx.x >> 5;
  const int mt = blockIdx.x;
  const int nt = blockIdx.y * 8 + wave;
  const int half = lane >> 4;
  const int ncol = lane & 15;

  const v16h* aF = (const v16h*)featfrag;
  const v16h* bF = (const v16h*)w1frag;

  v8f acc = {};
  #pragma unroll
  for (int s = 0; s < KSTEPS; ++s) {
    v16h a = aF[(mt * KSTEPS + s) * 32 + lane];
    v16h b = bF[(nt * KSTEPS + s) * 32 + lane];
    acc = __builtin_amdgcn_wmma_f32_16x16x32_f16(
        /*neg_a=*/false, a, /*neg_b=*/false, b,
        /*c_mod=*/(short)0, acc, /*reuse_a=*/false, /*reuse_b=*/false);
  }
  float bias = b1cat[nt * 16 + ncol];
  #pragma unroll
  for (int r = 0; r < 8; ++r) {
    int m = r + 8 * half;  // C/D layout: VGPR r holds M=r (lanes 0-15) / r+8 (16-31)
    hidden[(size_t)(mt * 16 + m) * NCOLS + nt * 16 + ncol] = fmaxf(acc[r] + bias, 0.f);
  }
}

// ---------------------------------------------------------------------------
// 4) Second layer (256 -> 8 outputs across 4 heads) + activations +
//    gaussian parameter precompute. One block per gaussian; wave w owns one
//    output scalar; lanes reduce 8 elements each via shfl_xor (wave32).
//    Output: params[row*8] = {px, py, conA, conB, conC, r, g, b}
// ---------------------------------------------------------------------------
__global__ __launch_bounds__(256)
void param_kernel(const float* __restrict__ hidden,
                  const float* off_w2, const float* off_b2,
                  const float* sc_w2,  const float* sc_b2,
                  const float* rot_w2, const float* rot_b2,
                  const float* col_w2, const float* col_b2,
                  const int* tH, const int* tW, float* __restrict__ params) {
  __shared__ float outv[8];
  const int row = blockIdx.x;
  const int wave = threadIdx.x >> 5, lane = threadIdx.x & 31;
  const int headOf[8] = {0, 0, 1, 1, 2, 3, 3, 3};
  const int colOf[8]  = {0, 1, 0, 1, 0, 0, 1, 2};
  const int odOf[4]   = {2, 2, 1, 3};
  const int h = headOf[wave], c = colOf[wave], od = odOf[h];
  const float* w2 = (h == 0) ? off_w2 : (h == 1) ? sc_w2 : (h == 2) ? rot_w2 : col_w2;
  const float* b2 = (h == 0) ? off_b2 : (h == 1) ? sc_b2 : (h == 2) ? rot_b2 : col_b2;

  const float* hid = hidden + (size_t)row * NCOLS + h * HID;
  float sum = 0.f;
  #pragma unroll
  for (int j = 0; j < 8; ++j) {
    int k = lane + 32 * j;
    sum += hid[k] * w2[k * od + c];
  }
  #pragma unroll
  for (int off = 16; off >= 1; off >>= 1) sum += __shfl_xor(sum, off, 32);
  if (lane == 0) outv[wave] = sum + b2[c];
  __syncthreads();

  if (threadIdx.x == 0) {
    float Htf = (float)tH[0], Wtf = (float)tW[0];
    float factor = fmaxf(Htf / (float)HI, Wtf / (float)WI);
    float off_scale = 3.f * (2.f * factor / fmaxf(Htf, Wtf));
    int n = row % NPB;
    float cy = -1.f + (2.f * (float)(n / WI) + 1.f) / (float)HI;
    float cx = -1.f + (2.f * (float)(n % WI) + 1.f) / (float)WI;
    float xy_y = cy + tanhf(outv[0]) * off_scale;
    float xy_x = cx + tanhf(outv[1]) * off_scale;
    float px = (xy_x + 1.f) * 0.5f * Wtf - 0.5f;
    float py = (xy_y + 1.f) * 0.5f * Htf - 0.5f;
    float s0 = (outv[2] > 0.f ? outv[2] : __expf(outv[2]) - 1.f) + 1.5f;
    float s1 = (outv[3] > 0.f ? outv[3] : __expf(outv[3]) - 1.f) + 1.5f;
    float rot = tanhf(outv[4]) * 3.14159265358979f;
    float cr = 1.f / (1.f + __expf(-outv[5]));
    float cg = 1.f / (1.f + __expf(-outv[6]));
    float cb = 1.f / (1.f + __expf(-outv[7]));
    float cs = __cosf(rot), sn = __sinf(rot);
    float sx2 = s0 * s0, sy2 = s1 * s1;
    float cva = cs * cs * sx2 + sn * sn * sy2;
    float cvb = cs * sn * (sx2 - sy2);
    float cvc = sn * sn * sx2 + cs * cs * sy2;
    float inv = 1.f / (cva * cvc - cvb * cvb);
    float* p = params + (size_t)row * 8;
    p[0] = px; p[1] = py;
    p[2] = cvc * inv;     // conA (multiplies dx^2)
    p[3] = -cvb * inv;    // conB (multiplies dx*dy)
    p[4] = cva * inv;     // conC (multiplies dy^2)
    p[5] = cr; p[6] = cg; p[7] = cb;
  }
}

// ---------------------------------------------------------------------------
// 5) Render: 64 pixels x 4 gaussian-slices per 256-thread block.
//    Gaussians staged in LDS (2 x float4 each), next chunk prefetched.
// ---------------------------------------------------------------------------
__global__ __launch_bounds__(256)
void render_kernel(const float* __restrict__ params, const int* tH, const int* tW,
                   float* __restrict__ out) {
  __shared__ float4 shA[CH];
  __shared__ float4 shB[CH];
  __shared__ float4 red[4][64];

  const int Wt = tW[0];
  const int tilesPerB = PIX / 64;
  const int b = blockIdx.x / tilesPerB;
  const int tile = blockIdx.x % tilesPerB;
  const int pixL = threadIdx.x & 63;
  const int slice = threadIdx.x >> 6;
  const int p = tile * 64 + pixL;
  const float gxf = (float)(p % Wt);
  const float gyf = (float)(p / Wt);

  const float4* gp = (const float4*)params + (size_t)b * NPB * 2;
  float accR = 0.f, accG = 0.f, accB = 0.f, accA = 0.f;

  for (int chunk = 0; chunk < NPB; chunk += CH) {
    __syncthreads();
    shA[threadIdx.x] = gp[(chunk + (int)threadIdx.x) * 2 + 0];
    shB[threadIdx.x] = gp[(chunk + (int)threadIdx.x) * 2 + 1];
    if (chunk + CH < NPB)
      __builtin_prefetch(&gp[(chunk + CH + (int)threadIdx.x) * 2], 0, 0);
    __syncthreads();
    #pragma unroll 4
    for (int j = 0; j < CH / 4; ++j) {
      int g = slice + 4 * j;
      float4 a = shA[g];   // px, py, conA, conB
      float4 bb = shB[g];  // conC, r, g, b
      float dx = gxf - a.x;
      float dy = gyf - a.y;
      float sigma = 0.5f * (a.z * dx * dx + bb.x * dy * dy) + a.w * dx * dy;
      float w = __expf(-sigma);
      float alpha = (sigma >= 0.f && w >= (1.f / 255.f)) ? fminf(w, 0.99f) : 0.f;
      accR += alpha * bb.y; accG += alpha * bb.z; accB += alpha * bb.w; accA += alpha;
    }
  }
  red[slice][pixL] = make_float4(accR, accG, accB, accA);
  __syncthreads();
  if (slice == 0) {
    float4 r0 = red[0][pixL], r1 = red[1][pixL], r2 = red[2][pixL], r3 = red[3][pixL];
    float R = r0.x + r1.x + r2.x + r3.x;
    float G = r0.y + r1.y + r2.y + r3.y;
    float Bc = r0.z + r1.z + r2.z + r3.z;
    float A = r0.w + r1.w + r2.w + r3.w;
    float T = fminf(fmaxf(1.f - A, 0.f), 1.f);
    float* o = out + ((size_t)b * PIX + p) * 3;
    o[0] = R + T; o[1] = G + T; o[2] = Bc + T;
  }
}

// ---------------------------------------------------------------------------
// Workspace layout (bytes, 256-aligned):
//   featfrag f16 [288][3][32][16]  @ 0          (884,736)
//   w1frag   f16 [64][3][32][16]   @ 884,736    (196,608)
//   b1cat    f32 [1024]            @ 1,081,344  (4,096)
//   hidden   f32 [4608 x 1024]     @ 1,085,440  (18,874,368)
//   params   f32 [4608 x 8]        @ 19,959,808 (147,456)   total ~20.1 MB
// ---------------------------------------------------------------------------
extern "C" void kernel_launch(void* const* d_in, const int* in_sizes, int n_in,
                              void* d_out, int out_size, void* d_ws, size_t ws_size,
                              hipStream_t stream) {
  const float* inp    = (const float*)d_in[0];
  const float* enc_w  = (const float*)d_in[1];
  const float* enc_b  = (const float*)d_in[2];
  const float* off_w1 = (const float*)d_in[3];
  const float* off_b1 = (const float*)d_in[4];
  const float* off_w2 = (const float*)d_in[5];
  const float* off_b2 = (const float*)d_in[6];
  const float* sc_w1  = (const float*)d_in[7];
  const float* sc_b1  = (const float*)d_in[8];
  const float* sc_w2  = (const float*)d_in[9];
  const float* sc_b2  = (const float*)d_in[10];
  const float* rot_w1 = (const float*)d_in[11];
  const float* rot_b1 = (const float*)d_in[12];
  const float* rot_w2 = (const float*)d_in[13];
  const float* rot_b2 = (const float*)d_in[14];
  const float* col_w1 = (const float*)d_in[15];
  const float* col_b1 = (const float*)d_in[16];
  const float* col_w2 = (const float*)d_in[17];
  const float* col_b2 = (const float*)d_in[18];
  const int* tH = (const int*)d_in[19];
  const int* tW = (const int*)d_in[20];

  char* ws = (char*)d_ws;
  _Float16* featfrag = (_Float16*)(ws);
  _Float16* w1frag   = (_Float16*)(ws + 884736);
  float*    b1cat    = (float*)(ws + 1081344);
  float*    hidden   = (float*)(ws + 1085440);
  float*    params   = (float*)(ws + 19959808);

  pack_w1_kernel<<<(KPAD * NCOLS + 255) / 256, 256, 0, stream>>>(
      off_w1, sc_w1, rot_w1, col_w1, off_b1, sc_b1, rot_b1, col_b1, w1frag, b1cat);
  conv_feat_kernel<<<(NROWS * KPAD + 255) / 256, 256, 0, stream>>>(
      inp, enc_w, enc_b, tH, tW, featfrag);
  mlp_gemm_kernel<<<dim3(MT, NT / 8), 256, 0, stream>>>(
      featfrag, w1frag, b1cat, hidden);
  param_kernel<<<NROWS, 256, 0, stream>>>(
      hidden, off_w2, off_b2, sc_w2, sc_b2, rot_w2, rot_b2, col_w2, col_b2, tH, tW, params);
  render_kernel<<<B_BATCH * (PIX / 64), 256, 0, stream>>>(params, tH, tW, (float*)d_out);
}
